// AttentionBlock_9612136809020
// MI455X (gfx1250) — compile-verified, matrix operand order checked by reading, not run on hardware
//
#include <hip/hip_runtime.h>
#include <hip/hip_bf16.h>

// ---------------- problem constants (from reference) ----------------
constexpr int Bc  = 2;
constexpr int Sc  = 2048;
constexpr int Dc  = 1024;
constexpr int Hc  = 16;
constexpr int HDc = 64;          // head dim

// ---------------- vector types ----------------
typedef __attribute__((ext_vector_type(16))) __bf16 v16bf;
typedef __attribute__((ext_vector_type(8)))  __bf16 bf16x8;
typedef __attribute__((ext_vector_type(4)))  __bf16 bf16x4;
typedef __attribute__((ext_vector_type(8)))  float  v8f;
typedef __attribute__((ext_vector_type(4)))  unsigned int u32x4;
typedef __attribute__((ext_vector_type(8)))  int    i32x8;
typedef __attribute__((ext_vector_type(4)))  int    i32x4;

// ---------------- CDNA5 async / TDM feature detection ----------------
#if defined(__has_builtin)
#  if __has_builtin(__builtin_amdgcn_global_load_async_to_lds_b128)
#    define USE_ASYNC_LDS 1
#  endif
#  if __has_builtin(__builtin_amdgcn_tensor_load_to_lds) && __has_builtin(__builtin_amdgcn_s_wait_tensorcnt)
#    define USE_TDM 1
#  endif
#endif

#if defined(USE_ASYNC_LDS)
// async 16B global->LDS copy, tracked by ASYNCcnt (no VGPR data round-trip)
__device__ inline void async_copy16(void* lds_dst, const void* gsrc) {
    __builtin_amdgcn_global_load_async_to_lds_b128(
        (__attribute__((address_space(1))) i32x4*)(void*)(gsrc),
        (__attribute__((address_space(3))) i32x4*)(lds_dst), 0, 0);
}
__device__ inline void wait_asynccnt0() {
#  if __has_builtin(__builtin_amdgcn_s_wait_asynccnt)
    __builtin_amdgcn_s_wait_asynccnt(0);
#  else
    asm volatile("s_wait_asynccnt 0x0" ::: "memory");
#  endif
}
#endif

#if defined(USE_TDM)
// One-shot 1-D TDM copy: nelem bf16 elements, contiguous global -> contiguous LDS.
// D# built per CDNA5 ISA (group0: count/type/addr, group1: dims).
__device__ inline void tdm_copy_1d_to_lds(const __bf16* gsrc, __bf16* ldst, unsigned nelem) {
    unsigned lds = (unsigned)(size_t)(__attribute__((address_space(3))) void*)ldst;
    unsigned long long ga = (unsigned long long)(size_t)gsrc;
    u32x4 g0;
    g0.x = 1u;                                                    // count=1, user desc
    g0.y = lds;                                                   // lds_addr (bytes)
    g0.z = (unsigned)ga;                                          // global_addr[31:0]
    g0.w = (unsigned)((ga >> 32) & 0x1FFFFFFull) | 0x80000000u;   // [56:32] | type=2
    i32x8 g1;
    g1[0] = 0x00010000;                                    // wg_mask=0, data_size=1 (2B)
    g1[1] = (int)((nelem & 0xFFFFu) << 16);                // tensor_dim0 lo16 @ [31:16]
    g1[2] = (int)(((nelem >> 16) & 0xFFFFu) | (1u << 16)); // dim0 hi16 | tensor_dim1=1
    g1[3] = (int)((nelem & 0xFFFFu) << 16);                // tile_dim0 @ [31:16]
    g1[4] = 0;                                             // tile_dim1=0, tile_dim2=0 (1-D)
    g1[5] = (int)nelem;                                    // tensor_dim0_stride lo32
    g1[6] = 0;
    g1[7] = 0;
    __builtin_amdgcn_tensor_load_to_lds(g0, g1,
        (i32x4){0,0,0,0}, (i32x4){0,0,0,0},
        (i32x8){0,0,0,0,0,0,0,0}, 0);
}
#endif

union AFrag { v16bf v; bf16x8 h[2]; };

// Load one 16x32 bf16 A/B fragment (row-major [16][>=32] with leading dim ld)
// per the CDNA5 ISA 16-bit A layout:
//   lanes 0-15 : row = lane,    K = {0..7, 16..23}
//   lanes 16-31: row = lane-16, K = {8..15, 24..31}
__device__ inline v16bf load_frag(const __bf16* base, int ld) {
    int lane = threadIdx.x & 31;
    int r    = lane & 15;
    int hf   = lane >> 4;
    const __bf16* p = base + r * ld + hf * 8;
    AFrag f;
    f.h[0] = *(const bf16x8*)(p);
    f.h[1] = *(const bf16x8*)(p + 16);
    return f.v;
}

__device__ inline v8f wmma_bf16(v16bf a, v16bf b, v8f c) {
    return __builtin_amdgcn_wmma_f32_16x16x32_bf16(
        false, a, false, b, (short)0, c, false, false);
}

// wave32-safe 16-lane-half reductions via ds_swizzle XOR patterns
__device__ inline float swz_max16(float v) {
    v = fmaxf(v, __int_as_float(__builtin_amdgcn_ds_swizzle(__float_as_int(v), 0x041f)));
    v = fmaxf(v, __int_as_float(__builtin_amdgcn_ds_swizzle(__float_as_int(v), 0x081f)));
    v = fmaxf(v, __int_as_float(__builtin_amdgcn_ds_swizzle(__float_as_int(v), 0x101f)));
    v = fmaxf(v, __int_as_float(__builtin_amdgcn_ds_swizzle(__float_as_int(v), 0x201f)));
    return v;
}
__device__ inline float swz_sum16(float v) {
    v += __int_as_float(__builtin_amdgcn_ds_swizzle(__float_as_int(v), 0x041f));
    v += __int_as_float(__builtin_amdgcn_ds_swizzle(__float_as_int(v), 0x081f));
    v += __int_as_float(__builtin_amdgcn_ds_swizzle(__float_as_int(v), 0x101f));
    v += __int_as_float(__builtin_amdgcn_ds_swizzle(__float_as_int(v), 0x201f));
    return v;
}

// ---------------- kernel 1: f32 -> bf16 (vectorized x4) ----------------
__global__ __launch_bounds__(256) void cvt_bf16_kernel(const float* __restrict__ in,
                                                       __bf16* __restrict__ out, int n4) {
    int i = blockIdx.x * 256 + threadIdx.x;
    if (i < n4) {
        float4 f = ((const float4*)in)[i];
        bf16x4 o = { (__bf16)f.x, (__bf16)f.y, (__bf16)f.z, (__bf16)f.w };
        ((bf16x4*)out)[i] = o;
    }
}

// stage one 128x32 bf16 tile of A and B into LDS (async DMA when available)
__device__ inline void stage_tiles(const __bf16* __restrict__ A, const __bf16* __restrict__ Bw,
                                   __bf16* sA, __bf16* sB,
                                   int m0, int n0, int k0, int Kd, int t) {
#pragma unroll
    for (int j = 0; j < 2; ++j) {
        int c   = t * 2 + j;
        int row = c >> 2;
        int col = (c & 3) * 8;
        const __bf16* ga = &A [(size_t)(m0 + row) * Kd + k0 + col];
        const __bf16* gb = &Bw[(size_t)(n0 + row) * Kd + k0 + col];
#if defined(USE_ASYNC_LDS)
        async_copy16(&sA[row * 32 + col], ga);
        async_copy16(&sB[row * 32 + col], gb);
#else
        *(bf16x8*)&sA[row * 32 + col] = *(const bf16x8*)ga;
        *(bf16x8*)&sB[row * 32 + col] = *(const bf16x8*)gb;
        __builtin_prefetch(ga + 32, 0, 3);
        __builtin_prefetch(gb + 32, 0, 3);
#endif
    }
}

// ---------------- kernel 2/5: WMMA GEMM  C[M,N] = A[M,K] * B[N,K]^T ----------------
// Double-buffered LDS pipeline: async-DMA tile k+1 while WMMA consumes tile k.
// MODE 0: write f32 C row-major [M,N]        (output projection)
// MODE 1: scatter bf16 into Q/K/V [B,H,S,64] (fused-QKV epilogue, head-major split)
template <int MODE>
__global__ __launch_bounds__(256) void gemm_bf16_kernel(
    const __bf16* __restrict__ A, const __bf16* __restrict__ Bw,
    float* __restrict__ C,
    __bf16* __restrict__ Qo, __bf16* __restrict__ Ko, __bf16* __restrict__ Vo,
    int M, int N, int Kd)
{
    __shared__ __bf16 sA[2][128 * 32];     // 2 x 8 KB
    __shared__ __bf16 sB[2][128 * 32];     // 2 x 8 KB

    const int t    = threadIdx.x;
    const int w    = t >> 5;
    const int lane = t & 31;
    const int wr   = w >> 2;        // wave M slot (0..1)  -> 64 rows each
    const int wc   = w & 3;         // wave N slot (0..3)  -> 32 cols each
    const int m0   = blockIdx.y * 128;
    const int n0   = blockIdx.x * 128;
    const int hf   = lane >> 4;
    const int ln   = lane & 15;

    v8f acc[4][2];
#pragma unroll
    for (int fm = 0; fm < 4; ++fm)
#pragma unroll
        for (int fn = 0; fn < 2; ++fn)
            acc[fm][fn] = (v8f){0.f,0.f,0.f,0.f,0.f,0.f,0.f,0.f};

    // prologue: stage tile 0
    stage_tiles(A, Bw, sA[0], sB[0], m0, n0, 0, Kd, t);
#if defined(USE_ASYNC_LDS)
    wait_asynccnt0();
#endif
    __syncthreads();

    for (int k0 = 0; k0 < Kd; k0 += 32) {
        const int cur = (k0 >> 5) & 1;
        // issue next tile's DMA into the alternate buffer (overlaps with WMMA below)
        if (k0 + 32 < Kd)
            stage_tiles(A, Bw, sA[cur ^ 1], sB[cur ^ 1], m0, n0, k0 + 32, Kd, t);

        v16bf af[4], bf_[2];
#pragma unroll
        for (int fm = 0; fm < 4; ++fm)
            af[fm] = load_frag(sA[cur] + (wr * 64 + fm * 16) * 32, 32);
#pragma unroll
        for (int fn = 0; fn < 2; ++fn)
            bf_[fn] = load_frag(sB[cur] + (wc * 32 + fn * 16) * 32, 32);
#pragma unroll
        for (int fm = 0; fm < 4; ++fm)
#pragma unroll
            for (int fn = 0; fn < 2; ++fn)
                acc[fm][fn] = wmma_bf16(af[fm], bf_[fn], acc[fm][fn]);

#if defined(USE_ASYNC_LDS)
        if (k0 + 32 < Kd) wait_asynccnt0();
#endif
        __syncthreads();   // next tile staged + everyone done reading cur
    }

    // epilogue: D-frag VGPR i holds (row = i + 8*half, col = lane&15)
#pragma unroll
    for (int fm = 0; fm < 4; ++fm)
#pragma unroll
        for (int fn = 0; fn < 2; ++fn)
#pragma unroll
            for (int i = 0; i < 8; ++i) {
                int m = m0 + wr * 64 + fm * 16 + i + 8 * hf;
                int n = n0 + wc * 32 + fn * 16 + ln;
                float val = acc[fm][fn][i];
                if (MODE == 0) {
                    C[(size_t)m * N + n] = val;
                } else {
                    int hh  = n / 192;
                    int rem = n - hh * 192;
                    int sel = rem >> 6;
                    int d   = rem & 63;
                    int b   = m >> 11;
                    int s   = m & (Sc - 1);
                    __bf16* dst = (sel == 0) ? Qo : (sel == 1) ? Ko : Vo;
                    dst[(((size_t)(b * Hc + hh)) * Sc + s) * HDc + d] = (__bf16)val;
                }
            }
}

// ---------------- kernel 3: rotate-half RoPE on Q,K (Q also scaled 1/sqrt(HD)) ----------------
__global__ __launch_bounds__(256) void rope_kernel(__bf16* __restrict__ Q,
                                                   __bf16* __restrict__ K,
                                                   const int* __restrict__ pos) {
    int t = blockIdx.x * 256 + threadIdx.x;   // B*H*S*32 threads
    int i = t & 31;
    int s = (t >> 5) & (Sc - 1);
    int h = (t >> 16) & (Hc - 1);
    int b = t >> 20;

    float p   = (float)pos[b * Sc + s];
    float inv = __expf(-(float)i * 0.28782313662425572f);   // 10000^(-i/32)
    float ang = p * inv;
    float c = __cosf(ang), sn = __sinf(ang);

    size_t base = (((size_t)(b * Hc + h)) * Sc + s) * HDc + i;
    float q1 = (float)Q[base], q2 = (float)Q[base + 32];
    float k1 = (float)K[base], k2 = (float)K[base + 32];
    const float qs = 0.125f;                  // 1/sqrt(64) folded into Q
    Q[base]      = (__bf16)((q1 * c - q2 * sn) * qs);
    Q[base + 32] = (__bf16)((q2 * c + q1 * sn) * qs);
    K[base]      = (__bf16)(k1 * c - k2 * sn);
    K[base + 32] = (__bf16)(k2 * c + k1 * sn);
}

// ---------------- kernel 4: causal flash attention, one (b,h,q-tile) per block ----------------
__global__ __launch_bounds__(256) void attn_kernel(
    const __bf16* __restrict__ Q, const __bf16* __restrict__ Kg,
    const __bf16* __restrict__ Vg, const int* __restrict__ pos,
    __bf16* __restrict__ ctx)
{
    __shared__ __bf16 sK [64 * 64];       // [kv][d]  -> B-frag layout for scores
    __shared__ __bf16 sVt[64 * 64];       // [d][kv]  -> B-frag layout for P*V
    __shared__ __bf16 sP [8 * 16 * 64];   // per-wave P scratch (D-frag -> A-frag)

    const int t    = threadIdx.x;
    const int w    = t >> 5;
    const int lane = t & 31;
    const int hf   = lane >> 4;
    const int ln   = lane & 15;

    const int bh = blockIdx.x;            // b*16 + h
    const int b  = bh >> 4;
    const int q0 = blockIdx.y * 128;
    const int qw = q0 + w * 16;           // this wave's first q row

    const __bf16* qbase = Q + (((size_t)bh) * Sc + qw) * HDc;
    v16bf qf0 = load_frag(qbase, HDc);
    v16bf qf1 = load_frag(qbase + 32, HDc);

    int pq[8];
#pragma unroll
    for (int i = 0; i < 8; ++i)
        pq[i] = pos[b * Sc + qw + i + 8 * hf];

    v8f o[4];
#pragma unroll
    for (int fn = 0; fn < 4; ++fn) o[fn] = (v8f){0.f,0.f,0.f,0.f,0.f,0.f,0.f,0.f};
    float mrow[8], lrow[8];
#pragma unroll
    for (int i = 0; i < 8; ++i) { mrow[i] = -3.0e38f; lrow[i] = 0.f; }

    const __bf16* kg  = Kg + ((size_t)bh) * Sc * HDc;
    const __bf16* vg  = Vg + ((size_t)bh) * Sc * HDc;
    __bf16*       sPw = sP + w * 16 * 64;

    for (int kv0 = 0; kv0 < q0 + 128; kv0 += 64) {
        // ---- stage K tile (contiguous 8KB): TDM if available, else per-lane loads ----
#if defined(USE_TDM)
        if (w == 0) {
            // one wave issues a single 1-D tensor DMA; TENSORcnt-tracked
            tdm_copy_1d_to_lds(kg + (size_t)kv0 * HDc, sK, 64 * HDc);
            __builtin_amdgcn_s_wait_tensorcnt(0);
        }
#endif
        {
            int row = t >> 2;             // 0..63 (kv)
            int c0  = (t & 3) * 16;       // 0,16,32,48 (d)
#if !defined(USE_TDM)
            const __bf16* kp = kg + (size_t)(kv0 + row) * HDc + c0;
            *(bf16x8*)&sK[row * 64 + c0]     = *(const bf16x8*)kp;
            *(bf16x8*)&sK[row * 64 + c0 + 8] = *(const bf16x8*)(kp + 8);
#endif
            // V tile transposed into [d][kv] so PV B-frags are two contiguous 16B reads
            const __bf16* vp = vg + (size_t)(kv0 + row) * HDc + c0;
            bf16x8 v0 = *(const bf16x8*)vp;
            bf16x8 v1 = *(const bf16x8*)(vp + 8);
#pragma unroll
            for (int j = 0; j < 8; ++j) {
                sVt[(c0 + j)     * 64 + row] = v0[j];
                sVt[(c0 + 8 + j) * 64 + row] = v1[j];
            }
        }
        __syncthreads();

        // ---- scores: S(16x64) = Q(16x64) * K^T ----
        v8f sc[4];
#pragma unroll
        for (int fn = 0; fn < 4; ++fn) {
            v8f z = (v8f){0.f,0.f,0.f,0.f,0.f,0.f,0.f,0.f};
            z = wmma_bf16(qf0, load_frag(sK + (fn * 16) * 64,      64), z);
            z = wmma_bf16(qf1, load_frag(sK + (fn * 16) * 64 + 32, 64), z);
            sc[fn] = z;
        }

        // ---- causal mask ----
#pragma unroll
        for (int fn = 0; fn < 4; ++fn)
#pragma unroll
            for (int i = 0; i < 8; ++i)
                if (kv0 + fn * 16 + ln > pq[i]) sc[fn][i] = -3.0e38f;

        // ---- online softmax ----
#pragma unroll
        for (int i = 0; i < 8; ++i) {
            float mx = fmaxf(fmaxf(sc[0][i], sc[1][i]), fmaxf(sc[2][i], sc[3][i]));
            mx = swz_max16(mx);
            float mnew  = fmaxf(mrow[i], mx);
            float scale = __expf(mrow[i] - mnew);
            mrow[i] = mnew;
            lrow[i] *= scale;
#pragma unroll
            for (int fn = 0; fn < 4; ++fn) o[fn][i] *= scale;
            float rs = 0.f;
#pragma unroll
            for (int fn = 0; fn < 4; ++fn) {
                float pexp = __expf(sc[fn][i] - mnew);
                sc[fn][i] = pexp;
                rs += pexp;
            }
            lrow[i] += swz_sum16(rs);
        }

        // ---- D-frag -> A-frag for P via per-wave LDS scratch ----
#pragma unroll
        for (int fn = 0; fn < 4; ++fn)
#pragma unroll
            for (int i = 0; i < 8; ++i)
                sPw[(i + 8 * hf) * 64 + fn * 16 + ln] = (__bf16)sc[fn][i];

        v16bf pf0 = load_frag(sPw, 64);
        v16bf pf1 = load_frag(sPw + 32, 64);

        // ---- O += P(16x64) * V(64x64) ----
#pragma unroll
        for (int fn = 0; fn < 4; ++fn) {
            o[fn] = wmma_bf16(pf0, load_frag(sVt + (fn * 16) * 64,      64), o[fn]);
            o[fn] = wmma_bf16(pf1, load_frag(sVt + (fn * 16) * 64 + 32, 64), o[fn]);
        }
        __syncthreads();   // protect sK/sVt before next tile
    }

    // ---- normalize and emit context in [B,S,H*HD] (bf16) ----
#pragma unroll
    for (int fn = 0; fn < 4; ++fn)
#pragma unroll
        for (int i = 0; i < 8; ++i) {
            int   qg  = qw + i + 8 * hf;
            float val = o[fn][i] / lrow[i];
            ctx[((size_t)b * Sc + qg) * Dc + (bh & 15) * HDc + fn * 16 + ln] = (__bf16)val;
        }
}

// ---------------- host-side launcher ----------------
extern "C" void kernel_launch(void* const* d_in, const int* in_sizes, int n_in,
                              void* d_out, int out_size, void* d_ws, size_t ws_size,
                              hipStream_t stream) {
    const float* x    = (const float*)d_in[0];   // [B,S,D]
    const int*   posp = (const int*)  d_in[1];   // [B,S]
    const float* Win  = (const float*)d_in[2];   // [3D,D]
    const float* Wout = (const float*)d_in[3];   // [D,D]
    float*       out  = (float*)d_out;           // [B,S,D]

    char* ws = (char*)d_ws;
    const size_t MB = 1u << 20;
    __bf16* Xbf    = (__bf16*)(ws);              //  8 MB  [4096,1024]
    __bf16* Winbf  = (__bf16*)(ws +  8 * MB);    //  6 MB  [3072,1024]
    __bf16* Woutbf = (__bf16*)(ws + 14 * MB);    //  2 MB  [1024,1024]
    __bf16* Qb     = (__bf16*)(ws + 16 * MB);    //  8 MB  [B,H,S,64]
    __bf16* Kb     = (__bf16*)(ws + 24 * MB);    //  8 MB
    __bf16* Vb     = (__bf16*)(ws + 32 * MB);    //  8 MB
    __bf16* Ctx    = (__bf16*)(ws + 40 * MB);    //  8 MB  [4096,1024]

    const int M = Bc * Sc;                       // 4096

    cvt_bf16_kernel<<<(M * Dc / 4 + 255) / 256, 256, 0, stream>>>(x, Xbf, M * Dc / 4);
    cvt_bf16_kernel<<<(3 * Dc * Dc / 4 + 255) / 256, 256, 0, stream>>>(Win, Winbf, 3 * Dc * Dc / 4);
    cvt_bf16_kernel<<<(Dc * Dc / 4 + 255) / 256, 256, 0, stream>>>(Wout, Woutbf, Dc * Dc / 4);

    gemm_bf16_kernel<1><<<dim3(3 * Dc / 128, M / 128), 256, 0, stream>>>(
        Xbf, Winbf, nullptr, Qb, Kb, Vb, M, 3 * Dc, Dc);

    rope_kernel<<<(Bc * Hc * Sc * 32) / 256, 256, 0, stream>>>(Qb, Kb, posp);

    attn_kernel<<<dim3(Bc * Hc, Sc / 128), 256, 0, stream>>>(Qb, Kb, Vb, posp, Ctx);

    gemm_bf16_kernel<0><<<dim3(Dc / 128, M / 128), 256, 0, stream>>>(
        Ctx, Woutbf, out, nullptr, nullptr, nullptr, M, Dc, Dc);
}
